// CSWinBlock_85864986182314
// MI455X (gfx1250) — matrix operand order repaired and checked by reading, not compiled, not run on hardware
//
#include <hip/hip_runtime.h>
#include <hip/hip_bf16.h>
#include <math.h>

// ---------------------------------------------------------------------------
// CSWin block for MI455X (gfx1250, wave32, WMMA).
// All dense GEMMs use v_wmma_f32_16x16x32_f16 with f16 operands, f32 acc.
// Fragment layouts follow CDNA5 ISA 7.12.2 (wave32):
//   A (16x32 f16): lane 0-15 -> M=lane (lanes 16-31: same M, K+8 split);
//                  per-lane f16 elements 0..7  = K khalf..khalf+7,
//                              elements 8..15 = K 16+khalf..23+khalf
//   B (32x16 f16): lane -> N=lane&15; per-lane elements e -> K as for A (e<8: khalf+e, e>=8: 8+khalf+e)
//   C/D (16x16 f32): lane -> N=lane&15; VGPR r -> M=r (+8 if lane>=16)
// Tiles are staged in LDS in *fragment-native* order so every fragment load is
// two contiguous 16B LDS loads; all fragment loads are hoisted ahead of the
// WMMA chain so a single s_wait_dscnt covers them and the WMMAs issue
// back-to-back on the matrix pipe.
// ---------------------------------------------------------------------------

typedef __attribute__((ext_vector_type(16))) _Float16 v16h;
typedef __attribute__((ext_vector_type(8)))  _Float16 v8h;
typedef __attribute__((ext_vector_type(8)))  float    v8f;

#define M_TOK   100352      // 32 * 56 * 56
#define RESO    56
#define NWIN    256         // 32 images * 8 windows
#define NPAD    416         // 392 padded to multiple of 32
#define NREAL   392
#define CHALF   64
#define HD      32

static __device__ inline v8f wmma32(v16h a, v16h b, v8f c) {
  return __builtin_amdgcn_wmma_f32_16x16x32_f16(
      /*neg_a=*/false, a, /*neg_b=*/false, b,
      /*c_mod=*/(short)0, c, /*reuse_a=*/false, /*reuse_b=*/false);
}

// Build a 16-element f16 fragment from two aligned contiguous 8xf16 chunks.
static __device__ inline v16h ld_frag2(const _Float16* p0, const _Float16* p1) {
  v8h lo = *(const v8h*)p0;
  v8h hi = *(const v8h*)p1;
  return __builtin_shufflevector(lo, hi, 0, 1, 2, 3, 4, 5, 6, 7,
                                 8, 9, 10, 11, 12, 13, 14, 15);
}

// A fragment from a row-major (M x 32) f16 tile, 16B-aligned rows.
static __device__ inline v16h frag_A_rows(const _Float16* tile, int lda,
                                          int mbase, int lane) {
  const _Float16* row = tile + (size_t)(mbase + (lane & 15)) * lda;
  int khalf = (lane >> 4) * 8;
  return ld_frag2(row + khalf, row + 16 + khalf);
}

// B fragment from an N-major (N x 32) transposed f16 tile.
static __device__ inline v16h frag_B_ncont(const _Float16* tileT, int nbase,
                                           int lane) {
  const _Float16* p = tileT + (size_t)(nbase + (lane & 15)) * 32 + (lane >> 4) * 8;
  return ld_frag2(p, p + 16);
}

// ---------------------------------------------------------------------------
// f32 -> f16 weight conversion
// ---------------------------------------------------------------------------
__global__ void k_f2h(const float* __restrict__ in, _Float16* __restrict__ out, int n) {
  int i = blockIdx.x * 256 + threadIdx.x;
  if (i < n) out[i] = (_Float16)in[i];
}

// ---------------------------------------------------------------------------
// LayerNorm over C=128, f32 in -> f16 out. One row per 128-thread block.
// ---------------------------------------------------------------------------
__global__ __launch_bounds__(128) void k_layernorm(
    const float* __restrict__ x, const float* __restrict__ g,
    const float* __restrict__ b, _Float16* __restrict__ out) {
  __shared__ float red[128];
  int row = blockIdx.x, c = threadIdx.x;
  float v = x[(size_t)row * 128 + c];
  red[c] = v; __syncthreads();
  for (int s = 64; s > 0; s >>= 1) { if (c < s) red[c] += red[c + s]; __syncthreads(); }
  float mu = red[0] * (1.f / 128.f); __syncthreads();
  float d = v - mu;
  red[c] = d * d; __syncthreads();
  for (int s = 64; s > 0; s >>= 1) { if (c < s) red[c] += red[c + s]; __syncthreads(); }
  float var = red[0] * (1.f / 128.f);
  out[(size_t)row * 128 + c] = (_Float16)(d * rsqrtf(var + 1e-5f) * g[c] + b[c]);
}

// ---------------------------------------------------------------------------
// Generic WMMA GEMM: C = A(MxK,f16) * B(KxN,f16) [+bias][GELU][+resid]
// Block = 256 threads (8 waves), tile 128x64, K step 32.
// A staged row-major (128x32); B staged TRANSPOSED (64 n-rows x 32 k).
// M % 128 == 0, N % 64 == 0, K % 32 == 0 (true for all uses here).
// ---------------------------------------------------------------------------
__global__ __launch_bounds__(256) void k_gemm_f16(
    const _Float16* __restrict__ A, const _Float16* __restrict__ B,
    const float* __restrict__ bias, const float* __restrict__ resid,
    float* __restrict__ outF, _Float16* __restrict__ outH,
    int M, int N, int K, int gelu) {
  __shared__ _Float16 As[128 * 32];    // row-major
  __shared__ _Float16 BsT[64 * 32];    // transposed: [n][k]
  int tid = threadIdx.x;
  int wave = tid >> 5, lane = tid & 31;
  int m0 = blockIdx.x * 128;
  int n0 = blockIdx.y * 64;

  int arow = tid >> 1, ahalf = tid & 1;  // A tile: 128 rows x (2 x 16)
  int brow = tid >> 3, bseg = tid & 7;   // B tile: 32 k-rows x (8 x 8 n)

  v8f acc[4];
#pragma unroll
  for (int j = 0; j < 4; ++j) acc[j] = {};

  for (int k0 = 0; k0 < K; k0 += 32) {
    const _Float16* ap = A + (size_t)(m0 + arow) * K + k0 + ahalf * 16;
    const _Float16* bp = B + (size_t)(k0 + brow) * N + n0 + bseg * 8;
    v8h a0 = *(const v8h*)ap;
    v8h a1 = *(const v8h*)(ap + 8);
    v8h bb = *(const v8h*)bp;
    if (k0 + 32 < K) {  // prefetch next K slice
      __builtin_prefetch(ap + 32, 0, 1);
      __builtin_prefetch(B + (size_t)(k0 + 32 + brow) * N + n0 + bseg * 8, 0, 1);
    }
    *(v8h*)&As[arow * 32 + ahalf * 16]     = a0;
    *(v8h*)&As[arow * 32 + ahalf * 16 + 8] = a1;
#pragma unroll
    for (int j = 0; j < 8; ++j)            // transpose scatter (stores only)
      BsT[(bseg * 8 + j) * 32 + brow] = bb[j];
    __syncthreads();

    // Hoist all fragment loads so one wait covers them, then issue the
    // four WMMAs back-to-back on the matrix pipe.
    v16h af  = frag_A_rows(As, 32, wave * 16, lane);
    v16h bf0 = frag_B_ncont(BsT, 0, lane);
    v16h bf1 = frag_B_ncont(BsT, 16, lane);
    v16h bf2 = frag_B_ncont(BsT, 32, lane);
    v16h bf3 = frag_B_ncont(BsT, 48, lane);
    acc[0] = wmma32(af, bf0, acc[0]);
    acc[1] = wmma32(af, bf1, acc[1]);
    acc[2] = wmma32(af, bf2, acc[2]);
    acc[3] = wmma32(af, bf3, acc[3]);
    __syncthreads();
  }

  int n_in = lane & 15;
  int mo = (lane >> 4) * 8;
#pragma unroll
  for (int j = 0; j < 4; ++j) {
    int gn = n0 + j * 16 + n_in;
    float bv = bias ? bias[gn] : 0.f;
#pragma unroll
    for (int r = 0; r < 8; ++r) {
      int gm = m0 + wave * 16 + mo + r;
      float v = acc[j][r] + bv;
      if (gelu) v = 0.5f * v * (1.f + erff(v * 0.70710678118654752f));
      if (resid) v += resid[(size_t)gm * N + gn];
      if (outF) outF[(size_t)gm * N + gn] = v;
      if (outH) outH[(size_t)gm * N + gn] = (_Float16)v;
    }
  }
}

// ---------------------------------------------------------------------------
// Window prep: img2windows + pos-add + focusing map, writes padded f16 q,k,v.
// grid (NPAD, NWIN), block 64 (channel). Norms are over the full 64 channels.
// ---------------------------------------------------------------------------
__global__ __launch_bounds__(64) void k_win_prep(
    const _Float16* __restrict__ qkv, const float* __restrict__ pos,
    const float* __restrict__ scale_p,
    _Float16* __restrict__ qw, _Float16* __restrict__ kw, _Float16* __restrict__ vw,
    int branch, int H_sp, int W_sp) {
  __shared__ float red[64];
  int c = threadIdx.x;
  int n = blockIdx.x;
  int win = blockIdx.y;
  size_t wbase = ((size_t)win * NPAD + n) * CHALF;
  if (n >= NREAL) {  // zero padding rows (uniform per block)
    qw[wbase + c] = (_Float16)0.f;
    kw[wbase + c] = (_Float16)0.f;
    vw[wbase + c] = (_Float16)0.f;
    return;
  }
  int nWw = RESO / W_sp, nWh = RESO / H_sp;
  int wpi = nWh * nWw;
  int bimg = win / wpi, wrem = win % wpi;
  int wh = wrem / nWw, wwi = wrem % nWw;
  int hs = n / W_sp, wsx = n % W_sp;
  int gl = (wh * H_sp + hs) * RESO + (wwi * W_sp + wsx);
  size_t row = ((size_t)bimg * (RESO * RESO) + gl) * 384;
  int ch = branch * CHALF + c;
  float q = (float)qkv[row + ch];
  float k = (float)qkv[row + 128 + ch];
  float v = (float)qkv[row + 256 + ch];
  k += pos[n * CHALF + c];
  float sc = log1pf(expf(scale_p[c]));  // softplus
  q = (fmaxf(q, 0.f) + 1e-6f) / sc;
  k = (fmaxf(k, 0.f) + 1e-6f) / sc;
  float qn, kn, q3n, k3n;
  float q3 = q * q * q, k3 = k * k * k;
#define REDUCE(val, dst)                                                    \
  red[c] = (val); __syncthreads();                                          \
  for (int s = 32; s > 0; s >>= 1) { if (c < s) red[c] += red[c + s]; __syncthreads(); } \
  dst = sqrtf(red[0]); __syncthreads();
  REDUCE(q * q, qn)
  REDUCE(k * k, kn)
  REDUCE(q3 * q3, q3n)
  REDUCE(k3 * k3, k3n)
#undef REDUCE
  qw[wbase + c] = (_Float16)(q3 / q3n * qn);
  kw[wbase + c] = (_Float16)(k3 / k3n * kn);
  vw[wbase + c] = (_Float16)v;
}

// ---------------------------------------------------------------------------
// ksum[wh][d] = sum_n k'[win][n][head*32+d]     grid 512, block 32
// ---------------------------------------------------------------------------
__global__ __launch_bounds__(32) void k_ksum(const _Float16* __restrict__ kw,
                                             float* __restrict__ ksum) {
  int whi = blockIdx.x, d = threadIdx.x;
  int win = whi >> 1, head = whi & 1;
  float s = 0.f;
  for (int n = 0; n < NREAL; ++n)
    s += (float)kw[((size_t)win * NPAD + n) * CHALF + head * HD + d];
  ksum[whi * HD + d] = s;
}

// ---------------------------------------------------------------------------
// kv[wh] (32x32) = k'^T (32xN) @ v (Nx32) via WMMA.  grid 512, block 128.
// Per K-step, k' and v tiles are transposed into LDS so fragment loads are
// contiguous ds_load_b128 pairs.
// ---------------------------------------------------------------------------
__global__ __launch_bounds__(128) void k_kv(const _Float16* __restrict__ kw,
                                            const _Float16* __restrict__ vw,
                                            float* __restrict__ kvbuf) {
  __shared__ _Float16 kT[32 * 32];  // [c][kk]
  __shared__ _Float16 vT[32 * 32];  // [c][kk]
  int whi = blockIdx.x;
  int win = whi >> 1, head = whi & 1;
  int wave = threadIdx.x >> 5, lane = threadIdx.x & 31;
  int mt = (wave >> 1) * 16;  // c tile
  int nt = (wave & 1) * 16;   // d tile
  const _Float16* kbase = kw + (size_t)win * NPAD * CHALF + head * HD;
  const _Float16* vbase = vw + (size_t)win * NPAD * CHALF + head * HD;
  int kkrow = threadIdx.x >> 2;  // 0..31
  int cseg = threadIdx.x & 3;    // 4 segs of 8 channels
  v8f acc = {};
  for (int k0 = 0; k0 < NPAD; k0 += 32) {
    v8h kv8 = *(const v8h*)(kbase + (size_t)(k0 + kkrow) * CHALF + cseg * 8);
    v8h vv8 = *(const v8h*)(vbase + (size_t)(k0 + kkrow) * CHALF + cseg * 8);
#pragma unroll
    for (int j = 0; j < 8; ++j) {
      kT[(cseg * 8 + j) * 32 + kkrow] = kv8[j];
      vT[(cseg * 8 + j) * 32 + kkrow] = vv8[j];
    }
    __syncthreads();
    v16h a = frag_A_rows(kT, 32, mt, lane);     // A[c][kk] = k'[kk][c]
    v16h b = frag_B_ncont(vT, nt, lane);        // B[kk][d]
    acc = wmma32(a, b, acc);
    __syncthreads();
  }
  int n = nt + (lane & 15);
  int mo = (lane >> 4) * 8;
#pragma unroll
  for (int r = 0; r < 8; ++r)
    kvbuf[((size_t)whi * 32 + mt + mo + r) * 32 + n] = acc[r];
}

// ---------------------------------------------------------------------------
// z[wh][n] = 1 / (q'[n] . ksum + 1e-6)    grid (512,13), block 32
// ---------------------------------------------------------------------------
__global__ __launch_bounds__(32) void k_z(const _Float16* __restrict__ qw,
                                          const float* __restrict__ ksum,
                                          float* __restrict__ z) {
  int whi = blockIdx.x;
  int n = blockIdx.y * 32 + threadIdx.x;
  int win = whi >> 1, head = whi & 1;
  float s = 0.f;
  for (int d = 0; d < HD; ++d)
    s += (float)qw[((size_t)win * NPAD + n) * CHALF + head * HD + d] * ksum[whi * HD + d];
  z[(size_t)whi * NPAD + n] = 1.f / (s + 1e-6f);
}

// ---------------------------------------------------------------------------
// fm = dwc5x5(v per head) + lepe3x3(v per window) + biases (f32)
// grid (392, 256), block 64 (channel)
// ---------------------------------------------------------------------------
__global__ __launch_bounds__(64) void k_fm(
    const _Float16* __restrict__ vw,
    const float* __restrict__ lepe_w, const float* __restrict__ lepe_b,
    const float* __restrict__ dwc_w, const float* __restrict__ dwc_b,
    float* __restrict__ fm, int H_sp, int W_sp) {
  int c = threadIdx.x;
  int n = blockIdx.x;
  int win = blockIdx.y;
  int hs = n / W_sp, wsx = n % W_sp;
  int d = c & 31;
  float acc = dwc_b[d] + lepe_b[c];
  for (int kh = 0; kh < 5; ++kh)
    for (int kw2 = 0; kw2 < 5; ++kw2) {
      int h = hs + kh - 2, w = wsx + kw2 - 2;
      float vv = (h < 0 || h >= H_sp || w < 0 || w >= W_sp)
                     ? 0.f
                     : (float)vw[((size_t)win * NPAD + h * W_sp + w) * CHALF + c];
      acc += dwc_w[(d * 5 + kh) * 5 + kw2] * vv;
    }
  for (int kh = 0; kh < 3; ++kh)
    for (int kw2 = 0; kw2 < 3; ++kw2) {
      int h = hs + kh - 1, w = wsx + kw2 - 1;
      float vv = (h < 0 || h >= H_sp || w < 0 || w >= W_sp)
                     ? 0.f
                     : (float)vw[((size_t)win * NPAD + h * W_sp + w) * CHALF + c];
      acc += lepe_w[(c * 3 + kh) * 3 + kw2] * vv;
    }
  fm[((size_t)win * NREAL + n) * CHALF + c] = acc;
}

// ---------------------------------------------------------------------------
// attn_out: X = (q'(Nx32) @ kv(32x32)) * z + fm, scattered windows2img into
// the merged (B, L, 128) f16 attention tensor. grid 512, block 256 (8 waves).
// kv is staged once into LDS as f16 transposed (fragment-native).
// ---------------------------------------------------------------------------
__global__ __launch_bounds__(256) void k_attn_out(
    const _Float16* __restrict__ qw, const float* __restrict__ kvbuf,
    const float* __restrict__ z, const float* __restrict__ fm,
    _Float16* __restrict__ attn, int branch, int H_sp, int W_sp) {
  __shared__ _Float16 kvs[32 * 32];  // [n][k]
  int whi = blockIdx.x;
  int win = whi >> 1, head = whi & 1;
  int wave = threadIdx.x >> 5, lane = threadIdx.x & 31;
  const _Float16* qb = qw + (size_t)win * NPAD * CHALF + head * HD;
  const float* kvb = kvbuf + (size_t)whi * 1024;
#pragma unroll
  for (int i = threadIdx.x; i < 1024; i += 256) {
    int kk = i >> 5, n = i & 31;
    kvs[n * 32 + kk] = (_Float16)kvb[i];
  }
  __syncthreads();
  int nWw = RESO / W_sp, nWh = RESO / H_sp;
  int wpi = nWh * nWw;
  int bimg = win / wpi, wrem = win % wpi;
  int whr = wrem / nWw, wwi = wrem % nWw;

  for (int t = wave; t < NPAD / 16; t += 8) {
    int mb = t * 16;
    // hoisted fragments, back-to-back WMMAs
    v16h a   = frag_A_rows(qb, CHALF, mb, lane);  // global_load_b128 pairs
    v16h bf0 = frag_B_ncont(kvs, 0, lane);
    v16h bf1 = frag_B_ncont(kvs, 16, lane);
    v8f acc0 = {}, acc1 = {};
    acc0 = wmma32(a, bf0, acc0);
    acc1 = wmma32(a, bf1, acc1);
#pragma unroll
    for (int j = 0; j < 2; ++j) {
      const v8f& acc = j ? acc1 : acc0;
      int n = j * 16 + (lane & 15);
      int mo = (lane >> 4) * 8;
#pragma unroll
      for (int r = 0; r < 8; ++r) {
        int mm = mb + mo + r;
        if (mm >= NREAL) continue;
        float val = acc[r] * z[(size_t)whi * NPAD + mm] +
                    fm[((size_t)win * NREAL + mm) * CHALF + head * HD + n];
        int hs = mm / W_sp, wsx = mm % W_sp;
        int gl = (whr * H_sp + hs) * RESO + (wwi * W_sp + wsx);
        attn[((size_t)bimg * (RESO * RESO) + gl) * 128 + branch * CHALF + head * HD + n] =
            (_Float16)val;
      }
    }
  }
}

// ---------------------------------------------------------------------------
// host launch
// ---------------------------------------------------------------------------
extern "C" void kernel_launch(void* const* d_in, const int* in_sizes, int n_in,
                              void* d_out, int out_size, void* d_ws, size_t ws_size,
                              hipStream_t stream) {
  const float* x        = (const float*)d_in[0];
  const float* n1g      = (const float*)d_in[1];
  const float* n1b      = (const float*)d_in[2];
  const float* qkv_w    = (const float*)d_in[3];
  const float* lepe_w0  = (const float*)d_in[4];
  const float* lepe_b0  = (const float*)d_in[5];
  const float* dwc_w0   = (const float*)d_in[6];
  const float* dwc_b0   = (const float*)d_in[7];
  const float* scale0   = (const float*)d_in[8];
  const float* pos0     = (const float*)d_in[9];
  const float* lepe_w1  = (const float*)d_in[10];
  const float* lepe_b1  = (const float*)d_in[11];
  const float* dwc_w1   = (const float*)d_in[12];
  const float* dwc_b1   = (const float*)d_in[13];
  const float* scale1   = (const float*)d_in[14];
  const float* pos1     = (const float*)d_in[15];
  const float* proj_w   = (const float*)d_in[16];
  const float* proj_b   = (const float*)d_in[17];
  const float* n2g      = (const float*)d_in[18];
  const float* n2b      = (const float*)d_in[19];
  const float* fc1_w    = (const float*)d_in[20];
  const float* fc1_b    = (const float*)d_in[21];
  const float* fc2_w    = (const float*)d_in[22];
  const float* fc2_b    = (const float*)d_in[23];
  float* out = (float*)d_out;

  char* ws = (char*)d_ws;
  size_t off = 0;
  auto carve = [&](size_t bytes) -> void* {
    void* p = ws + off;
    off += (bytes + 255) & ~(size_t)255;
    return p;
  };

  _Float16* XnH  = (_Float16*)carve((size_t)M_TOK * 128 * 2);
  _Float16* qkvH = (_Float16*)carve((size_t)M_TOK * 384 * 2);  // aliased by h1 later
  _Float16 *qwH[2], *kwH[2], *vwH[2];
  for (int b = 0; b < 2; ++b) {
    qwH[b] = (_Float16*)carve((size_t)NWIN * NPAD * CHALF * 2);
    kwH[b] = (_Float16*)carve((size_t)NWIN * NPAD * CHALF * 2);
    vwH[b] = (_Float16*)carve((size_t)NWIN * NPAD * CHALF * 2);
  }
  float *ksumB[2], *kvB[2], *zB[2], *fmB[2];
  for (int b = 0; b < 2; ++b) {
    ksumB[b] = (float*)carve((size_t)512 * HD * 4);
    kvB[b]   = (float*)carve((size_t)512 * 32 * 32 * 4);
    zB[b]    = (float*)carve((size_t)512 * NPAD * 4);
    fmB[b]   = (float*)carve((size_t)NWIN * NREAL * CHALF * 4);
  }
  _Float16* attnH = (_Float16*)carve((size_t)M_TOK * 128 * 2);
  float*    xres  = (float*)carve((size_t)M_TOK * 128 * 4);
  _Float16* y2H   = (_Float16*)carve((size_t)M_TOK * 128 * 2);
  _Float16* wqkvH = (_Float16*)carve((size_t)49152 * 2);
  _Float16* wprjH = (_Float16*)carve((size_t)16384 * 2);
  _Float16* wfc1H = (_Float16*)carve((size_t)65536 * 2);
  _Float16* wfc2H = (_Float16*)carve((size_t)65536 * 2);
  // fc1 output (M x 512 f16 = 103 MB) aliases the dead qkv/window region.
  _Float16* h1H = qkvH;

  // 1) weights -> f16
  k_f2h<<<(49152 + 255) / 256, 256, 0, stream>>>(qkv_w, wqkvH, 49152);
  k_f2h<<<(16384 + 255) / 256, 256, 0, stream>>>(proj_w, wprjH, 16384);
  k_f2h<<<(65536 + 255) / 256, 256, 0, stream>>>(fc1_w, wfc1H, 65536);
  k_f2h<<<(65536 + 255) / 256, 256, 0, stream>>>(fc2_w, wfc2H, 65536);

  // 2) LN1 -> f16
  k_layernorm<<<M_TOK, 128, 0, stream>>>(x, n1g, n1b, XnH);

  // 3) qkv = Xn @ qkv_w (f16 out)
  k_gemm_f16<<<dim3(M_TOK / 128, 384 / 64), 256, 0, stream>>>(
      XnH, wqkvH, nullptr, nullptr, nullptr, qkvH, M_TOK, 384, 128, 0);

  // 4) per-branch attention
  const float* pos[2] = {pos0, pos1};
  const float* scl[2] = {scale0, scale1};
  const float* lw[2]  = {lepe_w0, lepe_w1};
  const float* lb[2]  = {lepe_b0, lepe_b1};
  const float* dw[2]  = {dwc_w0, dwc_w1};
  const float* db[2]  = {dwc_b0, dwc_b1};
  const int Hsp[2] = {RESO, 7}, Wsp[2] = {7, RESO};
  for (int b = 0; b < 2; ++b) {
    k_win_prep<<<dim3(NPAD, NWIN), 64, 0, stream>>>(
        qkvH, pos[b], scl[b], qwH[b], kwH[b], vwH[b], b, Hsp[b], Wsp[b]);
    k_ksum<<<512, 32, 0, stream>>>(kwH[b], ksumB[b]);
    k_kv<<<512, 128, 0, stream>>>(kwH[b], vwH[b], kvB[b]);
    k_z<<<dim3(512, NPAD / 32), 32, 0, stream>>>(qwH[b], ksumB[b], zB[b]);
    k_fm<<<dim3(NREAL, NWIN), 64, 0, stream>>>(
        vwH[b], lw[b], lb[b], dw[b], db[b], fmB[b], Hsp[b], Wsp[b]);
    k_attn_out<<<512, 256, 0, stream>>>(
        qwH[b], kvB[b], zB[b], fmB[b], attnH, b, Hsp[b], Wsp[b]);
  }

  // 5) proj (+bias, +residual x) -> x_res (f32)
  k_gemm_f16<<<dim3(M_TOK / 128, 128 / 64), 256, 0, stream>>>(
      attnH, wprjH, proj_b, x, xres, nullptr, M_TOK, 128, 128, 0);

  // 6) LN2 -> f16
  k_layernorm<<<M_TOK, 128, 0, stream>>>(xres, n2g, n2b, y2H);

  // 7) fc1 (+bias, GELU) -> f16 (aliases qkv region)
  k_gemm_f16<<<dim3(M_TOK / 128, 512 / 64), 256, 0, stream>>>(
      y2H, wfc1H, fc1_b, nullptr, nullptr, h1H, M_TOK, 512, 128, 1);

  // 8) fc2 (+bias, +residual x_res) -> d_out (f32)
  k_gemm_f16<<<dim3(M_TOK / 128, 128 / 64), 256, 0, stream>>>(
      h1H, wfc2H, fc2_b, xres, out, nullptr, M_TOK, 128, 512, 0);
}
